// MHA_78030965834171
// MI455X (gfx1250) — compile-verified, matrix operand order checked by reading, not run on hardware
//
#include <hip/hip_runtime.h>
#include <stdint.h>

#define EMBED_DIM 1024
#define NUM_HEADS 16
#define HEAD_DIM  64
#define SEQ       2048
#define BATCH     2

typedef __attribute__((ext_vector_type(16))) __bf16 v16bf;
typedef __attribute__((ext_vector_type(8)))  __bf16 v8bf;
typedef __attribute__((ext_vector_type(8)))  float  v8f;
typedef __attribute__((ext_vector_type(4)))  unsigned int v4u;
typedef __attribute__((ext_vector_type(8)))  int v8i;
typedef __attribute__((ext_vector_type(4)))  int v4i;
typedef int b128_t __attribute__((vector_size(16)));   // matches builtin param

#define AS1 __attribute__((address_space(1)))
#define AS3 __attribute__((address_space(3)))

// ---- CDNA5 feature detection (compile-safe fallbacks) -----------------------

#if __has_builtin(__builtin_amdgcn_global_load_async_to_lds_b128) && \
    __has_builtin(__builtin_amdgcn_s_wait_asynccnt)
#define HAS_ASYNC 1
#else
#define HAS_ASYNC 0
#endif

#if __has_builtin(__builtin_amdgcn_tensor_load_to_lds) && \
    __has_builtin(__builtin_amdgcn_s_wait_tensorcnt)
#define HAS_TDM 1
#else
#define HAS_TDM 0
#endif

// ---------------------------------------------------------------- helpers ---

__device__ __forceinline__ __bf16 to_bf16(float f) {
  unsigned u = __builtin_bit_cast(unsigned, f);
  unsigned r = u + 0x7FFFu + ((u >> 16) & 1u);
  unsigned short h = (unsigned short)(r >> 16);
  return __builtin_bit_cast(__bf16, h);
}

__device__ __forceinline__ v8f zero8() {
  v8f z;
#pragma unroll
  for (int i = 0; i < 8; ++i) z[i] = 0.0f;
  return z;
}

__device__ __forceinline__ v8f wmma_bf16(v16bf a, v16bf b, v8f c) {
  return __builtin_amdgcn_wmma_f32_16x16x32_bf16(
      false, a, false, b, (short)0, c, false, false);
}

// 16B global -> LDS copy: CDNA5 async-to-LDS (ASYNCcnt) when available,
// else through-VGPR copy.
__device__ __forceinline__ void copy_b128_to_lds(const __bf16* g, __bf16* l) {
#if HAS_ASYNC
  __builtin_amdgcn_global_load_async_to_lds_b128(
      (AS1 b128_t*)(uintptr_t)g, (AS3 b128_t*)(unsigned)(uintptr_t)l, 0, 0);
#else
  *(uint4*)l = *(const uint4*)g;
#endif
}

template <int N> __device__ __forceinline__ void async_wait() {
#if HAS_ASYNC
  __builtin_amdgcn_s_wait_asynccnt(N);
#endif
}

// Load one 16x32 bf16 operand fragment (A tile, or B tile given as 16 rows of
// the transposed operand, lane <-> output column).
// CDNA5 16-bit layout: elements 0-7  <-> K = half*8 + 0..7
//                      elements 8-15 <-> K = half*8 + 16..23   (half = lane>>4)
__device__ __forceinline__ v16bf load_frag(const __bf16* __restrict__ base,
                                           int stride, int lane) {
  const int r  = lane & 15;
  const int hh = lane >> 4;
  const __bf16* p = base + (size_t)r * stride + hh * 8;
  v8bf lo = *(const v8bf*)(p);
  v8bf hi = *(const v8bf*)(p + 16);
  return __builtin_shufflevector(lo, hi, 0, 1, 2, 3, 4, 5, 6, 7,
                                 8, 9, 10, 11, 12, 13, 14, 15);
}

// ------------------------------------------------------------------ cast ----

__global__ __launch_bounds__(256)
void cast_bf16_kernel(const float* __restrict__ in, __bf16* __restrict__ out,
                      int n) {
  int i = blockIdx.x * 256 + threadIdx.x;
  if (i < n) out[i] = to_bf16(in[i]);
}

// ------------------------------------------------------------------ GEMM ----
// C[M,N] = A[M,K](bf16) @ W[N,K](bf16)^T + bias
// mode 0: out bf16 head layout [B,H,S,Hd]; mode 1: out f32 dense [M,N].
// 256 thr = 8 waves; block tile 256(M) x 128(N); waves 4(M)x2(N), each wave
// 64x64 = 16 WMMA accumulators. K staged 32 wide, double-buffered LDS fed by
// async-to-LDS; s_wait_asynccnt 6 overlaps next-slab DMA with current compute.

__global__ __launch_bounds__(256)
void gemm_bf16_kernel(const __bf16* __restrict__ A, const __bf16* __restrict__ W,
                      const float* __restrict__ bias, void* __restrict__ out,
                      int M, int N, int K, int mode) {
  __shared__ __align__(16) __bf16 As[2][256 * 40];   // 32 K + 8 pad per row
  __shared__ __align__(16) __bf16 Ws[2][128 * 40];

  const int tid  = threadIdx.x;
  const int lane = tid & 31;
  const int w    = tid >> 5;
  const int wm   = w & 3;        // 0..3 -> 64-row strip
  const int wn   = w >> 2;       // 0..1 -> 64-col strip
  const int m0   = blockIdx.y * 256;
  const int n0   = blockIdx.x * 128;

  v8f acc[4][4];
#pragma unroll
  for (int i = 0; i < 4; ++i)
#pragma unroll
    for (int j = 0; j < 4; ++j) acc[i][j] = zero8();

  auto stage = [&](int kb, int buf) {
#pragma unroll
    for (int i = 0; i < 4; ++i) {           // A: 256 rows x 4 segs
      int c = tid + 256 * i;
      int row = c >> 2, seg = c & 3;
      copy_b128_to_lds(&A[(size_t)(m0 + row) * K + kb + seg * 8],
                       &As[buf][row * 40 + seg * 8]);
    }
#pragma unroll
    for (int i = 0; i < 2; ++i) {           // W: 128 rows x 4 segs
      int c = tid + 256 * i;
      int row = c >> 2, seg = c & 3;
      copy_b128_to_lds(&W[(size_t)(n0 + row) * K + kb + seg * 8],
                       &Ws[buf][row * 40 + seg * 8]);
    }
  };

  const int nk = K >> 5;
  stage(0, 0);
  for (int ki = 0; ki < nk; ++ki) {
    const int buf = ki & 1;
    if (ki + 1 < nk) {
      stage((ki + 1) << 5, buf ^ 1);  // 6 async ops into the other buffer
      async_wait<6>();                // current buffer's 6 ops complete
    } else {
      async_wait<0>();
    }
    __syncthreads();

    v16bf af[4], bw[4];
#pragma unroll
    for (int t = 0; t < 4; ++t)
      af[t] = load_frag(&As[buf][(wm * 64 + t * 16) * 40], 40, lane);
#pragma unroll
    for (int t = 0; t < 4; ++t)
      bw[t] = load_frag(&Ws[buf][(wn * 64 + t * 16) * 40], 40, lane);
#pragma unroll
    for (int i = 0; i < 4; ++i)
#pragma unroll
      for (int j = 0; j < 4; ++j)
        acc[i][j] = wmma_bf16(af[i], bw[j], acc[i][j]);

    __syncthreads();  // all waves done reading buf before it is re-staged
  }

  const int hh = lane >> 4, ln = lane & 15;
#pragma unroll
  for (int i = 0; i < 4; ++i)
#pragma unroll
    for (int j = 0; j < 4; ++j)
#pragma unroll
      for (int e = 0; e < 8; ++e) {
        int m = m0 + wm * 64 + i * 16 + e + 8 * hh;
        int n = n0 + wn * 64 + j * 16 + ln;
        float val = acc[i][j][e] + bias[n];
        if (mode == 0) {
          int b = m / SEQ, s = m % SEQ;
          int h = n / HEAD_DIM, hd = n % HEAD_DIM;
          ((__bf16*)out)[(((size_t)b * NUM_HEADS + h) * SEQ + s) * HEAD_DIM + hd] =
              to_bf16(val);
        } else {
          ((float*)out)[(size_t)m * N + n] = val;
        }
      }
}

// -------------------------------------------------------- flash attention ---
// Q,K,V bf16 [B,H,S,Hd]; 256 thr = 8 waves; 128 q rows/block (16/wave, rows
// wave-local so softmax needs only half-wave shfl reductions). K tile is DMA'd
// by the Tensor Data Mover with hardware LDS padding (stride 72); V is staged
// transposed for the PV B-operand; P re-laid through per-wave LDS.

__global__ __launch_bounds__(256)
void flash_attn_kernel(const __bf16* __restrict__ Q, const __bf16* __restrict__ K,
                       const __bf16* __restrict__ V, __bf16* __restrict__ O) {
  __shared__ __align__(16) __bf16 Ks[64 * 72];       // [key][hd]  pad 8
  __shared__ __align__(16) __bf16 Vt[64 * 72];       // [hd][key]  pad 8
  __shared__ __align__(16) __bf16 Ps[8 * 16 * 72];   // per-wave P [16][64]

  const int tid  = threadIdx.x;
  const int lane = tid & 31;
  const int w    = tid >> 5;
  const int hh   = lane >> 4, ln = lane & 15;
  const int bh   = blockIdx.y;
  const int q0   = blockIdx.x * 128 + w * 16;

  const __bf16* Qp = Q + (size_t)bh * SEQ * HEAD_DIM;
  const __bf16* Kp = K + (size_t)bh * SEQ * HEAD_DIM;
  const __bf16* Vp = V + (size_t)bh * SEQ * HEAD_DIM;

  v16bf qf[2];
  qf[0] = load_frag(Qp + (size_t)q0 * HEAD_DIM, HEAD_DIM, lane);
  qf[1] = load_frag(Qp + (size_t)q0 * HEAD_DIM + 32, HEAD_DIM, lane);

  float m_run[8], l_run[8];
  v8f o[4];
#pragma unroll
  for (int j = 0; j < 8; ++j) { m_run[j] = -3.0e38f; l_run[j] = 0.0f; }
#pragma unroll
  for (int nt = 0; nt < 4; ++nt) o[nt] = zero8();

  __bf16* Pw = &Ps[w * 16 * 72];
  const float scale = 0.125f;  // 1/sqrt(64)

  for (int kb = 0; kb < SEQ; kb += 64) {
    __syncthreads();

    // ---- stage K tile [64 keys][64 hd] via TDM (padded rows), else lanes ---
#if HAS_TDM
    if (w == 0) {
      unsigned long long ga =
          (unsigned long long)(uintptr_t)(Kp + (size_t)kb * HEAD_DIM);
      unsigned lds = (unsigned)(uintptr_t)(&Ks[0]);  // LDS byte offset
      v4u g0;
      g0[0] = 1u;                                   // count=1 (valid D#)
      g0[1] = lds;                                  // lds_addr
      g0[2] = (unsigned)ga;                         // global_addr[31:0]
      g0[3] = (unsigned)((ga >> 32) & 0x1FFFFFFu)   // global_addr[56:32]
              | (2u << 30);                         // type=2 ("image")
      v8i g1;
      g1[0] = (int)((1u << 16)      // data_size = 2 bytes
                  | (1u << 20)      // pad_enable
                  | (4u << 22)      // pad_interval: 32 DWORDs (one 128B row)
                  | (3u << 25));    // pad_amount: 4 DWORDs (16B) -> stride 72
      g1[1] = (int)(64u << 16);     // tensor_dim0 = 64
      g1[2] = (int)(64u << 16);     // tensor_dim1 = 64
      g1[3] = (int)(64u << 16);     // tile_dim0 = 64
      g1[4] = (int)64u;             // tile_dim1 = 64
      g1[5] = (int)64u;             // tensor_dim0_stride = 64 (contiguous rows)
      g1[6] = 0;
      g1[7] = 0;
      v4i z4 = {0, 0, 0, 0};
#if defined(__clang_major__) && __clang_major__ >= 23
      v8i z8 = {0, 0, 0, 0, 0, 0, 0, 0};
      __builtin_amdgcn_tensor_load_to_lds(g0, g1, z4, z4, z8, 0);
#else
      __builtin_amdgcn_tensor_load_to_lds(g0, g1, z4, z4, 0);
#endif
      __builtin_amdgcn_s_wait_tensorcnt(0);
    }
#else
#pragma unroll
    for (int i = 0; i < 2; ++i) {
      int c = tid + 256 * i;
      int row = c >> 3, seg = c & 7;
      copy_b128_to_lds(&Kp[((size_t)kb + row) * HEAD_DIM + seg * 8],
                       &Ks[row * 72 + seg * 8]);
    }
#endif
    // ---- stage V transposed: Vt[hd][key] ----
#pragma unroll
    for (int i = 0; i < 2; ++i) {
      int c = tid + 256 * i;
      int key = c >> 3, hd0 = (c & 7) * 8;
      v8bf vv = *(const v8bf*)(&Vp[((size_t)kb + key) * HEAD_DIM + hd0]);
#pragma unroll
      for (int e = 0; e < 8; ++e) Vt[(hd0 + e) * 72 + key] = vv[e];
    }
    async_wait<0>();
    __syncthreads();

    // ---- scores: S[16 x 64] = Q x K^T ----
    v8f s[4];
#pragma unroll
    for (int t = 0; t < 4; ++t) s[t] = zero8();
#pragma unroll
    for (int kc = 0; kc < 2; ++kc)
#pragma unroll
      for (int t = 0; t < 4; ++t) {
        v16bf bk = load_frag(&Ks[(t * 16) * 72 + kc * 32], 72, lane);
        s[t] = wmma_bf16(qf[kc], bk, s[t]);
      }

    // ---- online softmax (D layout: elem j <-> row j+8*hh; lane&15 <-> col) -
    float mloc[8];
#pragma unroll
    for (int j = 0; j < 8; ++j) mloc[j] = -3.0e38f;
#pragma unroll
    for (int t = 0; t < 4; ++t)
#pragma unroll
      for (int j = 0; j < 8; ++j) {
        s[t][j] *= scale;
        mloc[j] = fmaxf(mloc[j], s[t][j]);
      }
#pragma unroll
    for (int mask = 8; mask >= 1; mask >>= 1)
#pragma unroll
      for (int j = 0; j < 8; ++j)
        mloc[j] = fmaxf(mloc[j], __shfl_xor(mloc[j], mask, 32));

    float ef[8];
#pragma unroll
    for (int j = 0; j < 8; ++j) {
      float mn = fmaxf(m_run[j], mloc[j]);
      ef[j] = __expf(m_run[j] - mn);
      m_run[j] = mn;
    }
#pragma unroll
    for (int nt = 0; nt < 4; ++nt)
#pragma unroll
      for (int j = 0; j < 8; ++j) o[nt][j] *= ef[j];

    float rs[8];
#pragma unroll
    for (int j = 0; j < 8; ++j) rs[j] = 0.0f;
#pragma unroll
    for (int t = 0; t < 4; ++t)
#pragma unroll
      for (int j = 0; j < 8; ++j) {
        float p = __expf(s[t][j] - m_run[j]);
        s[t][j] = p;
        rs[j] += p;
      }
#pragma unroll
    for (int mask = 8; mask >= 1; mask >>= 1)
#pragma unroll
      for (int j = 0; j < 8; ++j) rs[j] += __shfl_xor(rs[j], mask, 32);
#pragma unroll
    for (int j = 0; j < 8; ++j) l_run[j] = l_run[j] * ef[j] + rs[j];

    // ---- P through per-wave LDS into A-fragment layout ----
#pragma unroll
    for (int t = 0; t < 4; ++t)
#pragma unroll
      for (int j = 0; j < 8; ++j)
        Pw[(j + 8 * hh) * 72 + t * 16 + ln] = to_bf16(s[t][j]);

    // ---- O += P x V ----
#pragma unroll
    for (int kc = 0; kc < 2; ++kc) {
      v16bf ap = load_frag(Pw + kc * 32, 72, lane);
#pragma unroll
      for (int nt = 0; nt < 4; ++nt) {
        v16bf bv = load_frag(&Vt[(nt * 16) * 72 + kc * 32], 72, lane);
        o[nt] = wmma_bf16(ap, bv, o[nt]);
      }
    }
  }

  const int b = bh / NUM_HEADS, h = bh % NUM_HEADS;
#pragma unroll
  for (int nt = 0; nt < 4; ++nt)
#pragma unroll
    for (int j = 0; j < 8; ++j) {
      int row = q0 + j + 8 * hh;
      int col = h * HEAD_DIM + nt * 16 + ln;
      O[((size_t)b * SEQ + row) * EMBED_DIM + col] = to_bf16(o[nt][j] / l_run[j]);
    }
}

// ---------------------------------------------------------------- launch ----

extern "C" void kernel_launch(void* const* d_in, const int* in_sizes, int n_in,
                              void* d_out, int out_size, void* d_ws, size_t ws_size,
                              hipStream_t stream) {
  (void)in_sizes; (void)n_in; (void)out_size; (void)ws_size;
  const float* x  = (const float*)d_in[0];
  const float* Wq = (const float*)d_in[1];
  const float* bq = (const float*)d_in[2];
  const float* Wk = (const float*)d_in[3];
  const float* bk = (const float*)d_in[4];
  const float* Wv = (const float*)d_in[5];
  const float* bv = (const float*)d_in[6];
  const float* Wo = (const float*)d_in[7];
  const float* bo = (const float*)d_in[8];

  const int M = BATCH * SEQ;   // 4096
  const int D = EMBED_DIM;     // 1024
  const size_t szX = (size_t)M * D * sizeof(__bf16);
  const size_t szW = (size_t)D * D * sizeof(__bf16);
  const size_t szH = (size_t)BATCH * NUM_HEADS * SEQ * HEAD_DIM * sizeof(__bf16);

  char* ws = (char*)d_ws;
  size_t off = 0;
  __bf16* xb  = (__bf16*)(ws + off); off += szX;
  __bf16* Wqb = (__bf16*)(ws + off); off += szW;
  __bf16* Wkb = (__bf16*)(ws + off); off += szW;
  __bf16* Wvb = (__bf16*)(ws + off); off += szW;
  __bf16* Wob = (__bf16*)(ws + off); off += szW;
  __bf16* Qb  = (__bf16*)(ws + off); off += szH;
  __bf16* Kb  = (__bf16*)(ws + off); off += szH;
  __bf16* Vb  = (__bf16*)(ws + off); off += szH;
  __bf16* Ab  = (__bf16*)(ws + off); off += szX;

  int nx = M * D;
  cast_bf16_kernel<<<(nx + 255) / 256, 256, 0, stream>>>(x, xb, nx);
  int nw = D * D;
  cast_bf16_kernel<<<(nw + 255) / 256, 256, 0, stream>>>(Wq, Wqb, nw);
  cast_bf16_kernel<<<(nw + 255) / 256, 256, 0, stream>>>(Wk, Wkb, nw);
  cast_bf16_kernel<<<(nw + 255) / 256, 256, 0, stream>>>(Wv, Wvb, nw);
  cast_bf16_kernel<<<(nw + 255) / 256, 256, 0, stream>>>(Wo, Wob, nw);

  dim3 gg(D / 128, M / 256);  // (8, 16)
  gemm_bf16_kernel<<<gg, 256, 0, stream>>>(xb, Wqb, bq, Qb, M, D, D, 0);
  gemm_bf16_kernel<<<gg, 256, 0, stream>>>(xb, Wkb, bk, Kb, M, D, D, 0);
  gemm_bf16_kernel<<<gg, 256, 0, stream>>>(xb, Wvb, bv, Vb, M, D, D, 0);

  dim3 ga(SEQ / 128, BATCH * NUM_HEADS);  // (16, 32)
  flash_attn_kernel<<<ga, 256, 0, stream>>>(Qb, Kb, Vb, Ab);

  gemm_bf16_kernel<<<gg, 256, 0, stream>>>(Ab, Wob, bo, (float*)d_out, M, D, D, 1);
}
